// SelfAttentionLayer_6253472383148
// MI455X (gfx1250) — matrix operand order repaired and checked by reading, not compiled, not run on hardware
//
#include <hip/hip_runtime.h>

// ---------------------------------------------------------------------------
// Self-attention layer for MI455X (gfx1250, wave32, WMMA).
// B=2, T=2048, H=1024, 16 heads, head_size 64.  bf16 WMMA, f32 accumulate.
// GEMMs: 128x64 block tiles, K-step 64, LDS double-buffered with CDNA5
// GLOBAL_LOAD_ASYNC_TO_LDS_B128 (ASYNCcnt) overlapping compute.
// ---------------------------------------------------------------------------

typedef __bf16 bf16_t;
typedef __attribute__((ext_vector_type(16))) __bf16 v16bf;
typedef __attribute__((ext_vector_type(8)))  float  v8f;

union Frag16 { v16bf v; uint4 q[2]; };   // 16 bf16 = 32 bytes = 2 x uint4

#define WMMA_BF16(a, b, c) \
  __builtin_amdgcn_wmma_f32_16x16x32_bf16(false, (a), false, (b), (short)0, (c), false, false)

static constexpr int kB  = 2;
static constexpr int kT  = 2048;
static constexpr int kH  = 1024;
static constexpr int kNH = 16;
static constexpr int kHS = 64;
static constexpr int kBT = kB * kT;          // 4096 rows
static constexpr float kLn1e4_over32 = 0.28782313662425575f;  // ln(10000)/32

__device__ inline v8f v8f_zero() {
  v8f r;
#pragma unroll
  for (int i = 0; i < 8; ++i) r[i] = 0.0f;
  return r;
}

// ---------------------------------------------------------------------------
// CDNA5 async global->LDS copy (16B per lane), tracked by ASYNCcnt.
// vdst = per-lane LDS byte address (low 32 bits of the generic shared ptr),
// vaddr = 64-bit global address, GV addressing mode.
// ---------------------------------------------------------------------------
__device__ inline void async_copy_b128(bf16_t* lds, const bf16_t* g) {
  asm volatile("global_load_async_to_lds_b128 %0, %1, off"
               :
               : "v"((unsigned)(uintptr_t)lds), "v"(g)
               : "memory");
}
__device__ inline void wait_async0() {
  asm volatile("s_wait_asynccnt 0" ::: "memory");
}

// ---------------------------------------------------------------------------
// Elementwise f32 -> bf16 cast
// ---------------------------------------------------------------------------
__global__ void cast_f32_to_bf16(const float* __restrict__ in,
                                 bf16_t* __restrict__ out, int n) {
  int i = blockIdx.x * blockDim.x + threadIdx.x;
  if (i < n) out[i] = (bf16_t)in[i];
}

// ---------------------------------------------------------------------------
// GEMM core: 128(M) x 64(N) block tile per 256-thread block (8 waves).
// y[m][o] = sum_k A[m][k] * W[o][k]   (A: [M][1024] bf16, W: [1024][1024] bf16)
// Wave (wm = wave>>1, wn = wave&1): 32 M-rows x 32 N-cols = 4 C fragments,
// 8 WMMAs per 64-wide K step.  LDS double-buffered; async staging of buffer
// n+1 overlaps WMMA on buffer n; one barrier per K iteration.
// ---------------------------------------------------------------------------
static constexpr int kAsElems = 128 * 64;   // one A buffer
static constexpr int kWsElems = 64 * 64;    // one W buffer

__device__ inline void gemm_stage(const bf16_t* __restrict__ gA,
                                  const bf16_t* __restrict__ gW,
                                  bf16_t* As, bf16_t* Ws, int tid) {
  // A tile: 128 rows x 64 k = 1024 16B-chunks -> 4 per thread
#pragma unroll
  for (int i = 0; i < 4; ++i) {
    const int c   = tid + 256 * i;
    const int row = c >> 3, kc = (c & 7) << 3;
    async_copy_b128(As + row * 64 + kc, gA + (size_t)row * kH + kc);
  }
  // W tile: 64 rows x 64 k = 512 chunks -> 2 per thread
#pragma unroll
  for (int i = 0; i < 2; ++i) {
    const int c   = tid + 256 * i;
    const int row = c >> 3, kc = (c & 7) << 3;
    async_copy_b128(Ws + row * 64 + kc, gW + (size_t)row * kH + kc);
  }
}

__device__ inline void gemm_core_128x64(const bf16_t* __restrict__ A,
                                        const bf16_t* __restrict__ W,
                                        int mtile, int ntile,
                                        bf16_t* AsBuf, bf16_t* WsBuf,
                                        v8f c[2][2]) {
  const int tid  = threadIdx.x;
  const int lane = tid & 31;
  const int wave = tid >> 5;
  const int wm   = wave >> 1;     // 0..3 -> 32 M rows
  const int wn   = wave & 1;      // 0..1 -> 32 N cols

#pragma unroll
  for (int mi = 0; mi < 2; ++mi)
#pragma unroll
    for (int ns = 0; ns < 2; ++ns) c[mi][ns] = v8f_zero();

  const bf16_t* gA = A + (size_t)(mtile * 128) * kH;
  const bf16_t* gW = W + (size_t)(ntile * 64) * kH;

  gemm_stage(gA, gW, AsBuf, WsBuf, tid);          // prime buffer 0

  const int NI = kH / 64;                          // 16 iterations
  for (int it = 0; it < NI; ++it) {
    const int cur = it & 1;
    bf16_t* As = AsBuf + cur * kAsElems;
    bf16_t* Ws = WsBuf + cur * kWsElems;

    wait_async0();        // this wave's async copies (into buffer cur) done
    __syncthreads();      // all waves' copies done; prior reads of nxt done

    if (it + 1 < NI)      // overlap: stage next buffer while computing cur
      gemm_stage(gA + (it + 1) * 64, gW + (it + 1) * 64,
                 AsBuf + (1 - cur) * kAsElems, WsBuf + (1 - cur) * kWsElems,
                 tid);

    const int kb0 = (lane >> 4) << 4;   // 0 / 16  (A frag chunk base)
    const int bk  = (lane >> 4) << 4;   // 0 / 16  (B frag chunk base)
#pragma unroll
    for (int ks = 0; ks < 2; ++ks) {    // two 32-wide K slabs
      Frag16 fa[2];
#pragma unroll
      for (int mi = 0; mi < 2; ++mi) {
        const int arow = wm * 32 + mi * 16 + (lane & 15);
        fa[mi].q[0] = *(const uint4*)(As + arow * 64 + ks * 32 + kb0);
        fa[mi].q[1] = *(const uint4*)(As + arow * 64 + ks * 32 + kb0 + 8);
      }
#pragma unroll
      for (int ns = 0; ns < 2; ++ns) {
        const int bcol = wn * 32 + ns * 16 + (lane & 15);
        Frag16 fb;
        fb.q[0] = *(const uint4*)(Ws + bcol * 64 + ks * 32 + bk);
        fb.q[1] = *(const uint4*)(Ws + bcol * 64 + ks * 32 + bk + 8);
#pragma unroll
        for (int mi = 0; mi < 2; ++mi)
          c[mi][ns] = WMMA_BF16(fa[mi].v, fb.v, c[mi][ns]);
      }
    }
  }
}

// ---------------------------------------------------------------------------
// Fused QKV projection + bias + RoPE epilogue.
//  mode (blockIdx.z): 0 = Q (rope, store [b,h,t,d]), 1 = K (rope, [b,h,t,d]),
//                     2 = V (no rope, store transposed [b,h,d,t])
// RoPE (per reference): out[2i] = cos*u[2i] - sin*u[2i+1], out[2i+1] = 0,
//                       angle = (t+1) * 10000^(-i/32).
// ---------------------------------------------------------------------------
__global__ __launch_bounds__(256) void gemm_qkv(
    const bf16_t* __restrict__ A,
    const bf16_t* __restrict__ Wq, const bf16_t* __restrict__ Wk,
    const bf16_t* __restrict__ Wv,
    const float* __restrict__ bq, const float* __restrict__ bk,
    const float* __restrict__ bv,
    bf16_t* __restrict__ Qb, bf16_t* __restrict__ Kb,
    bf16_t* __restrict__ Vt) {
  __shared__ bf16_t AsBuf[2 * kAsElems];   // 32 KB
  __shared__ bf16_t WsBuf[2 * kWsElems];   // 16 KB

  const int mode = blockIdx.z;
  const bf16_t* W    = (mode == 0) ? Wq : (mode == 1) ? Wk : Wv;
  const float*  bias = (mode == 0) ? bq : (mode == 1) ? bk : bv;

  v8f c[2][2];
  gemm_core_128x64(A, W, blockIdx.x, blockIdx.y, AsBuf, WsBuf, c);

  const int lane = threadIdx.x & 31;
  const int wave = threadIdx.x >> 5;
  const int wm = wave >> 1, wn = wave & 1;
  const int hi = lane & 15;

#pragma unroll
  for (int mi = 0; mi < 2; ++mi) {
    const int rbase = blockIdx.x * 128 + wm * 32 + mi * 16 + ((lane >> 4) << 3);
#pragma unroll
    for (int ns = 0; ns < 2; ++ns) {
      const int o    = blockIdx.y * 64 + wn * 32 + ns * 16 + hi;
      const int head = o >> 6;
      const int d    = o & 63;
      const float bo = bias[o];
#pragma unroll
      for (int j = 0; j < 8; ++j) {
        const int m = rbase + j;
        const int b = m / kT, t = m % kT;
        float v = c[mi][ns][j] + bo;
        if (mode == 2) {
          Vt[(((size_t)b * kNH + head) * kHS + d) * kT + t] = (bf16_t)v;
        } else {
          const float vn = __shfl_xor(v, 1, 32);  // adjacent d (odd partner)
          float outv = 0.0f;
          if (!(d & 1)) {
            const int   i     = d >> 1;
            const float theta = __expf(-(float)i * kLn1e4_over32);
            const float ang   = (float)(t + 1) * theta;
            outv = cosf(ang) * v - sinf(ang) * vn;
          }
          bf16_t* dst = (mode == 0) ? Qb : Kb;
          dst[(((size_t)b * kNH + head) * kT + t) * kHS + d] = (bf16_t)outv;
        }
      }
    }
  }
}

// ---------------------------------------------------------------------------
// Output projection: y(bf16) x Wo^T + bo -> f32 out [b,t,h]
// ---------------------------------------------------------------------------
__global__ __launch_bounds__(256) void gemm_out(
    const bf16_t* __restrict__ A, const bf16_t* __restrict__ Wo,
    const float* __restrict__ bo, float* __restrict__ Out) {
  __shared__ bf16_t AsBuf[2 * kAsElems];
  __shared__ bf16_t WsBuf[2 * kWsElems];

  v8f c[2][2];
  gemm_core_128x64(A, Wo, blockIdx.x, blockIdx.y, AsBuf, WsBuf, c);

  const int lane = threadIdx.x & 31;
  const int wave = threadIdx.x >> 5;
  const int wm = wave >> 1, wn = wave & 1;

#pragma unroll
  for (int mi = 0; mi < 2; ++mi) {
    const int rbase = blockIdx.x * 128 + wm * 32 + mi * 16 + ((lane >> 4) << 3);
#pragma unroll
    for (int ns = 0; ns < 2; ++ns) {
      const int o = blockIdx.y * 64 + wn * 32 + ns * 16 + (lane & 15);
      const float bb = bo[o];
#pragma unroll
      for (int j = 0; j < 8; ++j) {
        const int m = rbase + j;
        Out[(size_t)m * kH + o] = c[mi][ns][j] + bb;
      }
    }
  }
}

// ---------------------------------------------------------------------------
// Flash attention, causal.  One block = one (b*h, q-tile of 64 rows).
// 128 threads = 4 waves; wave w owns q-rows [qt*64 + w*16, +16).
// Streams k-tiles of 64, online softmax kept in registers (rows spread across
// C-fragment layout; row reductions via __shfl_xor over the 16-lane halves).
// P re-fragmented A-layout through per-wave LDS.  Output y -> bf16 [b,t,h].
// ---------------------------------------------------------------------------
__global__ __launch_bounds__(128) void flash_attn(
    const bf16_t* __restrict__ Qb, const bf16_t* __restrict__ Kb,
    const bf16_t* __restrict__ Vt, bf16_t* __restrict__ Yb) {
  __shared__ bf16_t Qs[64 * 64];        // 8 KB
  __shared__ bf16_t Ps[4][16 * 64];     // 8 KB (per-wave P staging)

  const int tid  = threadIdx.x;
  const int lane = tid & 31;
  const int w    = tid >> 5;
  const int qt   = blockIdx.x;          // q tile, 0..31
  const int bh   = blockIdx.y;          // b*NH + head, 0..31

  const bf16_t* Qh = Qb + (size_t)bh * kT * kHS;
  const bf16_t* Kh = Kb + (size_t)bh * kT * kHS;
  const bf16_t* Vh = Vt + (size_t)bh * kHS * kT;

  // stage Q tile (64 x 64) in LDS via async copies
  for (int cch = tid; cch < 64 * 64 / 8; cch += 128) {
    const int row = cch >> 3, kc = (cch & 7) << 3;
    async_copy_b128(Qs + row * 64 + kc,
                    Qh + (size_t)(qt * 64 + row) * kHS + kc);
  }
  wait_async0();
  __syncthreads();

  // Q A-fragments for this wave (held for the whole loop)
  Frag16 aq[2];
  {
    const int row = w * 16 + (lane & 15);
    const int kb0 = (lane >> 4) << 4;
#pragma unroll
    for (int h2 = 0; h2 < 2; ++h2) {
      aq[h2].q[0] = *(const uint4*)(Qs + row * 64 + h2 * 32 + kb0);
      aq[h2].q[1] = *(const uint4*)(Qs + row * 64 + h2 * 32 + kb0 + 8);
    }
  }

  float mrow[8], lrow[8];
  v8f o[4];
#pragma unroll
  for (int j = 0; j < 8; ++j) { mrow[j] = -1e30f; lrow[j] = 0.0f; }
#pragma unroll
  for (int cs = 0; cs < 4; ++cs) o[cs] = v8f_zero();

  const int qrow0 = qt * 64 + w * 16 + ((lane >> 4) << 3);  // row of elem j
  const int hi    = lane & 15;

  for (int kt = 0; kt <= qt; ++kt) {
    const int kv0 = kt * 64;

    // ---- S = Q K^T (4 col-subtiles x 2 d-halves) ----
    v8f s[4];
#pragma unroll
    for (int cs = 0; cs < 4; ++cs) {
      v8f acc = v8f_zero();
      const int kv = kv0 + cs * 16 + hi;
#pragma unroll
      for (int h2 = 0; h2 < 2; ++h2) {
        Frag16 bk;
        const bf16_t* p = Kh + (size_t)kv * kHS + h2 * 32 + ((lane >> 4) << 4);
        bk.q[0] = *(const uint4*)p;
        bk.q[1] = *(const uint4*)(p + 8);
        acc = WMMA_BF16(aq[h2].v, bk.v, acc);
      }
      s[cs] = acc;
    }

    // ---- scale + causal mask (diagonal tile only) ----
    const bool diag = (kt == qt);
#pragma unroll
    for (int cs = 0; cs < 4; ++cs) {
#pragma unroll
      for (int j = 0; j < 8; ++j) {
        float v = s[cs][j] * 0.125f;           // 1/sqrt(64)
        if (diag && (kv0 + cs * 16 + hi) > (qrow0 + j)) v = -1e10f;
        s[cs][j] = v;
      }
    }

    // ---- online softmax: row max ----
    float vmax[8];
#pragma unroll
    for (int j = 0; j < 8; ++j) {
      float v = s[0][j];
      v = fmaxf(v, s[1][j]); v = fmaxf(v, s[2][j]); v = fmaxf(v, s[3][j]);
      vmax[j] = v;
    }
#pragma unroll
    for (int off = 1; off < 16; off <<= 1)
#pragma unroll
      for (int j = 0; j < 8; ++j)
        vmax[j] = fmaxf(vmax[j], __shfl_xor(vmax[j], off, 32));

    float alpha[8];
#pragma unroll
    for (int j = 0; j < 8; ++j) {
      const float mn = fmaxf(mrow[j], vmax[j]);
      alpha[j] = __expf(mrow[j] - mn);
      mrow[j]  = mn;
    }

    // ---- p = exp(s - m), row sum ----
    float rs[8];
#pragma unroll
    for (int j = 0; j < 8; ++j) rs[j] = 0.0f;
#pragma unroll
    for (int cs = 0; cs < 4; ++cs)
#pragma unroll
      for (int j = 0; j < 8; ++j) {
        const float p = __expf(s[cs][j] - mrow[j]);
        s[cs][j] = p;
        rs[j] += p;
      }
#pragma unroll
    for (int off = 1; off < 16; off <<= 1)
#pragma unroll
      for (int j = 0; j < 8; ++j) rs[j] += __shfl_xor(rs[j], off, 32);
#pragma unroll
    for (int j = 0; j < 8; ++j) lrow[j] = lrow[j] * alpha[j] + rs[j];
#pragma unroll
    for (int cs = 0; cs < 4; ++cs)
#pragma unroll
      for (int j = 0; j < 8; ++j) o[cs][j] *= alpha[j];

    // ---- stage P (C-layout) -> LDS -> A-layout fragments ----
    __syncthreads();   // uniform trip count across all 4 waves
#pragma unroll
    for (int cs = 0; cs < 4; ++cs)
#pragma unroll
      for (int j = 0; j < 8; ++j) {
        const int prow = j + ((lane >> 4) << 3);
        Ps[w][prow * 64 + cs * 16 + hi] = (bf16_t)s[cs][j];
      }
    __syncthreads();

    // ---- O += P V  (V pre-transposed: consecutive t at fixed d) ----
    {
      const int prow = lane & 15;
      const int kb0  = (lane >> 4) << 4;
#pragma unroll
      for (int h2 = 0; h2 < 2; ++h2) {
        Frag16 ap;
        ap.q[0] = *(const uint4*)(&Ps[w][prow * 64 + h2 * 32 + kb0]);
        ap.q[1] = *(const uint4*)(&Ps[w][prow * 64 + h2 * 32 + kb0 + 8]);
#pragma unroll
        for (int cs = 0; cs < 4; ++cs) {
          Frag16 bv;
          const bf16_t* p =
              Vh + (size_t)(cs * 16 + hi) * kT + kv0 + h2 * 32 + ((lane >> 4) << 4);
          bv.q[0] = *(const uint4*)p;
          bv.q[1] = *(const uint4*)(p + 8);
          o[cs] = WMMA_BF16(ap.v, bv.v, o[cs]);
        }
      }
    }
  }

  // ---- epilogue: y = O / l, store [b, t, h] ----
  const int b = bh >> 4, head = bh & 15;
#pragma unroll
  for (int cs = 0; cs < 4; ++cs)
#pragma unroll
    for (int j = 0; j < 8; ++j) {
      const int t = qrow0 + j;
      const int d = cs * 16 + hi;
      const float val = o[cs][j] / lrow[j];
      Yb[((size_t)b * kT + t) * kH + head * kHS + d] = (bf16_t)val;
    }
}

// ---------------------------------------------------------------------------
// Launch
// ---------------------------------------------------------------------------
extern "C" void kernel_launch(void* const* d_in, const int* in_sizes, int n_in,
                              void* d_out, int out_size, void* d_ws, size_t ws_size,
                              hipStream_t stream) {
  const float* x  = (const float*)d_in[0];
  const float* Wq = (const float*)d_in[1];
  const float* bq = (const float*)d_in[2];
  const float* Wk = (const float*)d_in[3];
  const float* bk = (const float*)d_in[4];
  const float* Wv = (const float*)d_in[5];
  const float* bv = (const float*)d_in[6];
  const float* Wo = (const float*)d_in[7];
  const float* bo = (const float*)d_in[8];
  float* out = (float*)d_out;

  char* ws = (char*)d_ws;
  size_t off = 0;
  auto alloc = [&](size_t bytes) {
    void* p = ws + off;
    off += (bytes + 255) & ~(size_t)255;
    return p;
  };
  bf16_t* xb  = (bf16_t*)alloc((size_t)kBT * kH * 2);
  bf16_t* wqb = (bf16_t*)alloc((size_t)kH * kH * 2);
  bf16_t* wkb = (bf16_t*)alloc((size_t)kH * kH * 2);
  bf16_t* wvb = (bf16_t*)alloc((size_t)kH * kH * 2);
  bf16_t* wob = (bf16_t*)alloc((size_t)kH * kH * 2);
  bf16_t* Qb  = (bf16_t*)alloc((size_t)kBT * kH * 2);
  bf16_t* Kb  = (bf16_t*)alloc((size_t)kBT * kH * 2);
  bf16_t* Vt  = (bf16_t*)alloc((size_t)kBT * kH * 2);
  bf16_t* yb  = (bf16_t*)alloc((size_t)kBT * kH * 2);

  // casts
  {
    const int nx = kBT * kH;
    cast_f32_to_bf16<<<(nx + 255) / 256, 256, 0, stream>>>(x, xb, nx);
    const int nw = kH * kH;
    cast_f32_to_bf16<<<(nw + 255) / 256, 256, 0, stream>>>(Wq, wqb, nw);
    cast_f32_to_bf16<<<(nw + 255) / 256, 256, 0, stream>>>(Wk, wkb, nw);
    cast_f32_to_bf16<<<(nw + 255) / 256, 256, 0, stream>>>(Wv, wvb, nw);
    cast_f32_to_bf16<<<(nw + 255) / 256, 256, 0, stream>>>(Wo, wob, nw);
  }

  // fused QKV projection + bias + RoPE
  {
    dim3 grid(kBT / 128, kH / 64, 3);
    gemm_qkv<<<grid, 256, 0, stream>>>(xb, wqb, wkb, wvb, bq, bk, bv, Qb, Kb, Vt);
  }

  // causal flash attention
  {
    dim3 grid(kT / 64, kB * kNH);
    flash_attn<<<grid, 128, 0, stream>>>(Qb, Kb, Vt, yb);
  }

  // output projection -> f32
  {
    dim3 grid(kBT / 128, kH / 64);
    gemm_out<<<grid, 256, 0, stream>>>(yb, wob, bo, out);
  }
}
